// MultiheadAttention_56650618634402
// MI455X (gfx1250) — matrix operand order repaired and checked by reading, not compiled, not run on hardware
//
#include <hip/hip_runtime.h>

typedef __bf16 bf16_t;
typedef __attribute__((ext_vector_type(4)))  bf16_t v4bf;
typedef __attribute__((ext_vector_type(8)))  bf16_t v8bf;
typedef __attribute__((ext_vector_type(16))) bf16_t v16bf;
typedef __attribute__((ext_vector_type(8)))  float  v8f;

#define EMB   768
#define HEADS 12
#define HDIM  64
#define BATCH 4
#define SEQ   2048
#define MTOT  (BATCH*SEQ)   /* 8192 */

static __device__ __forceinline__ bf16_t f2bf(float f) { return (bf16_t)f; }

static __device__ __forceinline__ v16bf join8(v8bf lo, v8bf hi) {
  return __builtin_shufflevector(lo, hi, 0,1,2,3,4,5,6,7,8,9,10,11,12,13,14,15);
}

static __device__ __forceinline__ v8f wmma_bf16(v16bf a, v16bf b, v8f c) {
  return __builtin_amdgcn_wmma_f32_16x16x32_bf16(false, a, false, b, (short)0, c, false, false);
}

/* Async DMA: global -> LDS, 16 bytes per lane, tracked by ASYNCcnt (no VGPR data). */
static __device__ __forceinline__ void async_ld16(const bf16_t* lds_dst, const bf16_t* g_src) {
  unsigned lds_off = (unsigned)(size_t)lds_dst;   /* low 32 bits of generic ptr = LDS addr */
  asm volatile("global_load_async_to_lds_b128 %0, %1, off"
               :: "v"(lds_off), "v"(g_src) : "memory");
}
static __device__ __forceinline__ void wait_async0() {
  asm volatile("s_wait_asynccnt 0x0" ::: "memory");
}

/* ---------------- fp32 -> bf16 conversion ---------------- */
__global__ __launch_bounds__(256) void cvt_f32_bf16(const float* __restrict__ src,
                                                    bf16_t* __restrict__ dst, int n4) {
  int i = blockIdx.x * 256 + threadIdx.x;
  if (i >= n4) return;
  float4 f = ((const float4*)src)[i];
  v4bf o; o[0] = f2bf(f.x); o[1] = f2bf(f.y); o[2] = f2bf(f.z); o[3] = f2bf(f.w);
  ((v4bf*)dst)[i] = o;
}

/* ---------------- fused QKV projection ----------------
 * C[m,n] = sum_k x[m,k] * W[n,k] + b[n]
 * grid.x = 18 (3 weights x 6 N-tiles of 128), grid.y = 64 (M tiles of 128)
 * Q,K stored [B,H,S,64] bf16 ; V stored transposed [B,H,64,S] bf16
 * Async double-buffered LDS staging.
 */
__global__ __launch_bounds__(256) void gemm_qkv(
    const bf16_t* __restrict__ xa,
    const bf16_t* __restrict__ wq, const bf16_t* __restrict__ wk, const bf16_t* __restrict__ wv,
    const float* __restrict__ bq, const float* __restrict__ bk, const float* __restrict__ bv,
    bf16_t* __restrict__ qo, bf16_t* __restrict__ ko, bf16_t* __restrict__ vto)
{
  __shared__ __attribute__((aligned(16))) bf16_t As[2][128 * 40];  /* pad 32->40 */
  __shared__ __attribute__((aligned(16))) bf16_t Bs[2][128 * 40];
  const int tid  = threadIdx.x;
  const int lane = tid & 31, wave = tid >> 5;
  const int lq   = lane & 15, half = lane >> 4;
  const int wm   = wave & 3,  wn   = wave >> 2;      /* 4 x 2 wave grid */
  const int ntg  = blockIdx.x;
  const int which = ntg / 6;
  const int n0   = (ntg % 6) * 128;
  const int m0   = blockIdx.y * 128;
  const bf16_t* wb  = (which == 0) ? wq : (which == 1) ? wk : wv;
  const float*  bia = (which == 0) ? bq : (which == 1) ? bk : bv;

  /* each thread DMAs 2 16B chunks of A and of B per tile */
  const int r0 = tid >> 2,         c0 = (tid & 3) << 3;          /* chunk tid      */
  const int r1 = (tid + 256) >> 2, c1 = ((tid + 256) & 3) << 3;  /* chunk tid+256  */

  auto stage = [&](int buf, int k0) {
    async_ld16(&As[buf][r0 * 40 + c0], &xa[(size_t)(m0 + r0) * EMB + k0 + c0]);
    async_ld16(&As[buf][r1 * 40 + c1], &xa[(size_t)(m0 + r1) * EMB + k0 + c1]);
    async_ld16(&Bs[buf][r0 * 40 + c0], &wb[(size_t)(n0 + r0) * EMB + k0 + c0]);
    async_ld16(&Bs[buf][r1 * 40 + c1], &wb[(size_t)(n0 + r1) * EMB + k0 + c1]);
  };

  v8f acc[2][4];
  v8f z = {};
#pragma unroll
  for (int mt = 0; mt < 2; ++mt)
#pragma unroll
    for (int nt = 0; nt < 4; ++nt) acc[mt][nt] = z;

  stage(0, 0);
  wait_async0();
  __syncthreads();

  for (int ks = 0; ks < EMB / 32; ++ks) {
    const int buf = ks & 1;
    if (ks + 1 < EMB / 32) stage(buf ^ 1, (ks + 1) * 32);   /* DMA next tile during compute */

    v16bf a[2], b[4];
#pragma unroll
    for (int mt = 0; mt < 2; ++mt) {
      int row = wm * 32 + mt * 16 + lq;
      a[mt] = join8(*(v8bf*)&As[buf][row * 40 + 8 * half],
                    *(v8bf*)&As[buf][row * 40 + 16 + 8 * half]);
    }
#pragma unroll
    for (int nt = 0; nt < 4; ++nt) {
      int row = wn * 64 + nt * 16 + lq;
      b[nt] = join8(*(v8bf*)&Bs[buf][row * 40 + 8 * half],
                    *(v8bf*)&Bs[buf][row * 40 + 16 + 8 * half]);
    }
#pragma unroll
    for (int mt = 0; mt < 2; ++mt)
#pragma unroll
      for (int nt = 0; nt < 4; ++nt)
        acc[mt][nt] = wmma_bf16(a[mt], b[nt], acc[mt][nt]);

    wait_async0();
    __syncthreads();
  }

  const int bidx = m0 >> 11;                      /* 2048 rows per batch */
#pragma unroll
  for (int mt = 0; mt < 2; ++mt) {
    int s_base = (m0 & 2047) + wm * 32 + mt * 16 + 8 * half;
#pragma unroll
    for (int nt = 0; nt < 4; ++nt) {
      int   ng = n0 + wn * 64 + nt * 16 + lq;
      float bb = bia[ng];
      int h = ng >> 6, d = ng & 63;
      if (which == 2) {                           /* V transposed: contiguous in s */
        v8bf pk;
#pragma unroll
        for (int r = 0; r < 8; ++r) pk[r] = f2bf(acc[mt][nt][r] + bb);
        *(v8bf*)&vto[((size_t)(bidx * HEADS + h) * HDIM + d) * SEQ + s_base] = pk;
      } else {
        bf16_t* base = ((which == 0) ? qo : ko) +
                       (((size_t)(bidx * HEADS + h) * SEQ + s_base) * HDIM + d);
#pragma unroll
        for (int r = 0; r < 8; ++r) base[(size_t)r * HDIM] = f2bf(acc[mt][nt][r] + bb);
      }
    }
  }
}

/* ---------------- flash attention ----------------
 * grid.x = B*H*16 ; each wave: 16 queries ; key chunks of 64, async double-buffered LDS.
 * Computes S^T = K*Q^T, online softmax, O^T += V^T * P^T.
 */
__global__ __launch_bounds__(256) void attn(const bf16_t* __restrict__ qb_,
                                            const bf16_t* __restrict__ kb_,
                                            const bf16_t* __restrict__ vtb_,
                                            bf16_t* __restrict__ ob_)
{
  __shared__ __attribute__((aligned(16))) bf16_t Ks [2][64 * 72];   /* pad 64->72 */
  __shared__ __attribute__((aligned(16))) bf16_t Vts[2][64 * 72];
  const int tid  = threadIdx.x, lane = tid & 31, wave = tid >> 5;
  const int lq   = lane & 15, half = lane >> 4;
  const bool h0  = (half == 0);
  const int bh   = blockIdx.x >> 4;
  const int q0   = (blockIdx.x & 15) * 128 + wave * 16;

  const int r0 = tid >> 3,         c0 = (tid & 7) << 3;
  const int r1 = (tid + 256) >> 3, c1 = ((tid + 256) & 7) << 3;

  auto stage = [&](int buf, int kc) {
    async_ld16(&Ks [buf][r0 * 72 + c0], &kb_ [((size_t)bh * SEQ  + kc + r0) * HDIM + c0]);
    async_ld16(&Ks [buf][r1 * 72 + c1], &kb_ [((size_t)bh * SEQ  + kc + r1) * HDIM + c1]);
    async_ld16(&Vts[buf][r0 * 72 + c0], &vtb_[((size_t)bh * HDIM + r0) * SEQ + kc + c0]);
    async_ld16(&Vts[buf][r1 * 72 + c1], &vtb_[((size_t)bh * HDIM + r1) * SEQ + kc + c1]);
  };

  /* Q as B-operand: lane = query col, elems = head dims */
  const bf16_t* qrow = qb_ + ((size_t)bh * SEQ + q0 + lq) * HDIM;
  v16bf qf[2];
#pragma unroll
  for (int kw = 0; kw < 2; ++kw) {
    const bf16_t* p = qrow + kw * 32 + 8 * half;
    qf[kw] = join8(*(const v8bf*)p, *(const v8bf*)(p + 16));
  }

  v8f z = {};
  v8f o[4];
#pragma unroll
  for (int dt = 0; dt < 4; ++dt) o[dt] = z;
  float run_m = -INFINITY, run_l = 0.f;

  stage(0, 0);
  wait_async0();
  __syncthreads();

  for (int kc = 0; kc < SEQ; kc += 64) {
    const int buf = (kc >> 6) & 1;
    if (kc + 64 < SEQ) stage(buf ^ 1, kc + 64);

    /* S^T tiles: M = key rows, N = query cols */
    v8f st[4];
#pragma unroll
    for (int kt = 0; kt < 4; ++kt) st[kt] = z;
#pragma unroll
    for (int kt = 0; kt < 4; ++kt) {
      int row = kt * 16 + lq;
#pragma unroll
      for (int kw = 0; kw < 2; ++kw) {
        v16bf af = join8(*(v8bf*)&Ks[buf][row * 72 + kw * 32 + 8 * half],
                         *(v8bf*)&Ks[buf][row * 72 + kw * 32 + 16 + 8 * half]);
        st[kt] = wmma_bf16(af, qf[kw], st[kt]);
      }
    }

    /* scale + online softmax (axis = key rows: VGPR dim + half swap) */
    float cmax = -INFINITY;
#pragma unroll
    for (int t = 0; t < 4; ++t)
#pragma unroll
      for (int r = 0; r < 8; ++r) { float v = st[t][r] * 0.125f; st[t][r] = v; cmax = fmaxf(cmax, v); }
    cmax = fmaxf(cmax, __shfl_xor(cmax, 16));
    float nm   = fmaxf(run_m, cmax);
    float corr = __expf(run_m - nm);
    run_m = nm;
    float csum = 0.f;
#pragma unroll
    for (int t = 0; t < 4; ++t)
#pragma unroll
      for (int r = 0; r < 8; ++r) { float e = __expf(st[t][r] - nm); st[t][r] = e; csum += e; }
    csum += __shfl_xor(csum, 16);
    run_l = run_l * corr + csum;
#pragma unroll
    for (int dt = 0; dt < 4; ++dt)
#pragma unroll
      for (int r = 0; r < 8; ++r) o[dt][r] *= corr;

    /* P^T as B-operand: per-lane cross-half exchange, no LDS transpose */
    v16bf pb[2];
#pragma unroll
    for (int kb = 0; kb < 2; ++kb) {
      int t0 = 2 * kb, t1 = 2 * kb + 1;
      v16bf f;
#pragma unroll
      for (int e = 0; e < 8; ++e) {
        float oth = __shfl_xor(st[t1][e], 16);
        f[e] = f2bf(h0 ? st[t0][e] : oth);
      }
#pragma unroll
      for (int e = 0; e < 8; ++e) {
        float oth = __shfl_xor(st[t0][e], 16);
        f[8 + e] = f2bf(h0 ? oth : st[t1][e]);
      }
      pb[kb] = f;
    }

    /* O^T += V^T * P^T */
#pragma unroll
    for (int dt = 0; dt < 4; ++dt) {
      int row = dt * 16 + lq;
#pragma unroll
      for (int kb = 0; kb < 2; ++kb) {
        v16bf af = join8(*(v8bf*)&Vts[buf][row * 72 + kb * 32 + 16 * half],
                         *(v8bf*)&Vts[buf][row * 72 + kb * 32 + 16 * half + 8]);
        o[dt] = wmma_bf16(af, pb[kb], o[dt]);
      }
    }

    wait_async0();
    __syncthreads();
  }

  float inv = 1.f / run_l;
  int b = bh / HEADS, h = bh % HEADS;
  int s = q0 + lq;
  bf16_t* op = ob_ + (size_t)(b * SEQ + s) * EMB + h * HDIM;
#pragma unroll
  for (int dt = 0; dt < 4; ++dt) {
    v8bf pk;
#pragma unroll
    for (int r = 0; r < 8; ++r) pk[r] = f2bf(o[dt][r] * inv);
    *(v8bf*)(op + dt * 16 + 8 * half) = pk;
  }
}

/* ---------------- output projection (bf16 A, fp32 out) ---------------- */
__global__ __launch_bounds__(256) void gemm_out(const bf16_t* __restrict__ xa,
                                                const bf16_t* __restrict__ wb,
                                                const float* __restrict__ bia,
                                                float* __restrict__ out)
{
  __shared__ __attribute__((aligned(16))) bf16_t As[2][128 * 40];
  __shared__ __attribute__((aligned(16))) bf16_t Bs[2][128 * 40];
  const int tid  = threadIdx.x;
  const int lane = tid & 31, wave = tid >> 5;
  const int lq   = lane & 15, half = lane >> 4;
  const int wm   = wave & 3,  wn   = wave >> 2;
  const int n0   = blockIdx.x * 128;
  const int m0   = blockIdx.y * 128;

  const int r0 = tid >> 2,         c0 = (tid & 3) << 3;
  const int r1 = (tid + 256) >> 2, c1 = ((tid + 256) & 3) << 3;

  auto stage = [&](int buf, int k0) {
    async_ld16(&As[buf][r0 * 40 + c0], &xa[(size_t)(m0 + r0) * EMB + k0 + c0]);
    async_ld16(&As[buf][r1 * 40 + c1], &xa[(size_t)(m0 + r1) * EMB + k0 + c1]);
    async_ld16(&Bs[buf][r0 * 40 + c0], &wb[(size_t)(n0 + r0) * EMB + k0 + c0]);
    async_ld16(&Bs[buf][r1 * 40 + c1], &wb[(size_t)(n0 + r1) * EMB + k0 + c1]);
  };

  v8f acc[2][4];
  v8f z = {};
#pragma unroll
  for (int mt = 0; mt < 2; ++mt)
#pragma unroll
    for (int nt = 0; nt < 4; ++nt) acc[mt][nt] = z;

  stage(0, 0);
  wait_async0();
  __syncthreads();

  for (int ks = 0; ks < EMB / 32; ++ks) {
    const int buf = ks & 1;
    if (ks + 1 < EMB / 32) stage(buf ^ 1, (ks + 1) * 32);

    v16bf a[2], b[4];
#pragma unroll
    for (int mt = 0; mt < 2; ++mt) {
      int row = wm * 32 + mt * 16 + lq;
      a[mt] = join8(*(v8bf*)&As[buf][row * 40 + 8 * half],
                    *(v8bf*)&As[buf][row * 40 + 16 + 8 * half]);
    }
#pragma unroll
    for (int nt = 0; nt < 4; ++nt) {
      int row = wn * 64 + nt * 16 + lq;
      b[nt] = join8(*(v8bf*)&Bs[buf][row * 40 + 8 * half],
                    *(v8bf*)&Bs[buf][row * 40 + 16 + 8 * half]);
    }
#pragma unroll
    for (int mt = 0; mt < 2; ++mt)
#pragma unroll
      for (int nt = 0; nt < 4; ++nt)
        acc[mt][nt] = wmma_bf16(a[mt], b[nt], acc[mt][nt]);

    wait_async0();
    __syncthreads();
  }

#pragma unroll
  for (int mt = 0; mt < 2; ++mt) {
    int mrow = m0 + wm * 32 + mt * 16 + 8 * half;
#pragma unroll
    for (int nt = 0; nt < 4; ++nt) {
      int   ng = n0 + wn * 64 + nt * 16 + lq;
      float bb = bia[ng];
      float* p = out + (size_t)mrow * EMB + ng;
#pragma unroll
      for (int r = 0; r < 8; ++r) p[(size_t)r * EMB] = acc[mt][nt][r] + bb;
    }
  }
}

/* ---------------- host launcher ---------------- */
extern "C" void kernel_launch(void* const* d_in, const int* in_sizes, int n_in,
                              void* d_out, int out_size, void* d_ws, size_t ws_size,
                              hipStream_t stream) {
  const float* x    = (const float*)d_in[0];
  const float* wq_w = (const float*)d_in[1];
  const float* wq_b = (const float*)d_in[2];
  const float* wk_w = (const float*)d_in[3];
  const float* wk_b = (const float*)d_in[4];
  const float* wv_w = (const float*)d_in[5];
  const float* wv_b = (const float*)d_in[6];
  const float* wo_w = (const float*)d_in[7];
  const float* wo_b = (const float*)d_in[8];

  char*  ws  = (char*)d_ws;
  size_t off = 0;
  auto alloc = [&](size_t bytes) -> void* {
    void* p = ws + off;
    off = (off + bytes + 255) & ~(size_t)255;
    return p;
  };
  const size_t XE = (size_t)MTOT * EMB;                  /* 8192*768  */
  const size_t QE = (size_t)BATCH * HEADS * SEQ * HDIM;  /* 6291456   */

  bf16_t* xb  = (bf16_t*)alloc(XE * 2);
  bf16_t* wqb = (bf16_t*)alloc((size_t)EMB * EMB * 2);
  bf16_t* wkb = (bf16_t*)alloc((size_t)EMB * EMB * 2);
  bf16_t* wvb = (bf16_t*)alloc((size_t)EMB * EMB * 2);
  bf16_t* wob = (bf16_t*)alloc((size_t)EMB * EMB * 2);
  bf16_t* q   = (bf16_t*)alloc(QE * 2);
  bf16_t* k   = (bf16_t*)alloc(QE * 2);
  bf16_t* vt  = (bf16_t*)alloc(QE * 2);
  bf16_t* ob  = (bf16_t*)alloc(XE * 2);

  int n4x = (int)(XE / 4);
  int n4w = (EMB * EMB) / 4;
  cvt_f32_bf16<<<(n4x + 255) / 256, 256, 0, stream>>>(x, xb, n4x);
  cvt_f32_bf16<<<(n4w + 255) / 256, 256, 0, stream>>>(wq_w, wqb, n4w);
  cvt_f32_bf16<<<(n4w + 255) / 256, 256, 0, stream>>>(wk_w, wkb, n4w);
  cvt_f32_bf16<<<(n4w + 255) / 256, 256, 0, stream>>>(wv_w, wvb, n4w);
  cvt_f32_bf16<<<(n4w + 255) / 256, 256, 0, stream>>>(wo_w, wob, n4w);

  gemm_qkv<<<dim3(18, MTOT / 128), 256, 0, stream>>>(xb, wqb, wkb, wvb,
                                                     wq_b, wk_b, wv_b, q, k, vt);
  attn<<<dim3(BATCH * HEADS * (SEQ / 128)), 256, 0, stream>>>(q, k, vt, ob);
  gemm_out<<<dim3(EMB / 128, MTOT / 128), 256, 0, stream>>>(ob, wob, wo_b, (float*)d_out);
}
